// VisualBranch_ican_84610855731244
// MI455X (gfx1250) — matrix operand order repaired and checked by reading, not compiled, not run on hardware
//
#include <hip/hip_runtime.h>
#include <hip/hip_bf16.h>
#include <math.h>
#include <stdint.h>

// ---------------------------------------------------------------------------
// CDNA5 (gfx1250) WMMA types
// ---------------------------------------------------------------------------
typedef __attribute__((ext_vector_type(16))) __bf16          v16bf;
typedef __attribute__((ext_vector_type(16))) unsigned short  v16us;
typedef __attribute__((ext_vector_type(8)))  float           v8f;

// Problem constants (from reference)
#define NB   32
#define NK   16
#define NR   25
#define NC   512
#define NHW  2304
#define NOBJ 512
#define DIN  2048
#define DOUT 1024
#define DCAT 3072

// Compiler-level memory clobber: orders LDS tile re-reads after a TDM write.
#define LDS_TOUCH() asm volatile("" ::: "memory")

// ---------------------------------------------------------------------------
// Tensor Data Mover availability (builtin arity differs across toolchains)
// ---------------------------------------------------------------------------
#if defined(__has_builtin)
#  if __has_builtin(__builtin_amdgcn_tensor_load_to_lds)
#    define HAVE_TDM 1
#  else
#    define HAVE_TDM 0
#  endif
#else
#  define HAVE_TDM 0
#endif

#if HAVE_TDM
// Opaque never-executed store: the TDM builtin writes LDS invisibly to alias
// analysis; without any IR store to the tile the compiler folds tile loads to
// undef and deletes them.  An s_mov-sourced opaque zero guarding a store with
// a dynamic index makes every tile load live, at zero runtime cost (uniform
// scalar branch, never taken).
#define LDS_SEED(tb, mask)                                                      \
    {                                                                           \
        int _z;                                                                 \
        asm volatile("s_mov_b32 %0, 0" : "=s"(_z));                             \
        if (_z) tb[0][_z & (mask)] = (unsigned short)_z;                        \
    }
#else
#define LDS_SEED(tb, mask)
#endif

#if HAVE_TDM
typedef __attribute__((ext_vector_type(4))) unsigned int tdm_v4u;
typedef __attribute__((ext_vector_type(8))) int          tdm_v8i;
typedef __attribute__((ext_vector_type(4))) int          tdm_v4i;

// 2-D bf16 tile load Global->LDS via TDM.  D# per CDNA5 ISA §8.3-8.5.
static __device__ __forceinline__ void
tdm_load_2d_bf16(unsigned lds_off, const void* gptr,
                 unsigned tile_d0, unsigned tile_d1,
                 unsigned tensor_d0, unsigned tensor_d1,
                 unsigned long long stride_elems) {
    unsigned long long ga = (unsigned long long)(uintptr_t)gptr;
    tdm_v4u g0;
    g0.x = 1u;                                         // count=1 (valid user D#)
    g0.y = lds_off;                                    // lds_addr (bytes)
    g0.z = (unsigned)(ga & 0xFFFFFFFFu);               // global_addr[31:0]
    g0.w = (unsigned)((ga >> 32) & 0x01FFFFFFu)        // global_addr[56:32]
         | (2u << 30);                                 // type=2 ("image")
    tdm_v8i g1;
    g1[0] = (int)(1u << 16);                           // wg_mask=0, data_size=1 (2B)
    g1[1] = (int)((tensor_d0 & 0xFFFFu) << 16);        // tensor_dim0[15:0] @ bit48
    g1[2] = (int)(((tensor_d0 >> 16) & 0xFFFFu)
         |  ((tensor_d1 & 0xFFFFu) << 16));            // tensor_dim1[15:0]
    g1[3] = (int)(((tensor_d1 >> 16) & 0xFFFFu)
         |  ((tile_d0 & 0xFFFFu) << 16));              // tile_dim0
    g1[4] = (int)(tile_d1 & 0xFFFFu);                  // tile_dim1 (tile_dim2=0)
    g1[5] = (int)(unsigned)(stride_elems & 0xFFFFFFFFu);
    g1[6] = (int)(unsigned)((stride_elems >> 32) & 0xFFFFu);
    g1[7] = 0;
    tdm_v4i z4 = {0, 0, 0, 0};
#if defined(__clang_major__) && (__clang_major__ >= 23)
    tdm_v8i z8 = {0, 0, 0, 0, 0, 0, 0, 0};
    __builtin_amdgcn_tensor_load_to_lds(g0, g1, z4, z4, z8, 0);
#else
    __builtin_amdgcn_tensor_load_to_lds(g0, g1, z4, z4, 0);
#endif
}
#endif // HAVE_TDM

// fp32 -> bf16 round-to-nearest-even
static __device__ __forceinline__ unsigned bfr(float f) {
    union { float f; unsigned u; } x; x.f = f;
    return (x.u + 0x7FFFu + ((x.u >> 16) & 1u)) >> 16;
}
static __device__ __forceinline__ unsigned short f2bf(float f) {
    return (unsigned short)bfr(f);
}

// ---------------------------------------------------------------------------
// K0: bulk fp32 -> bf16, float4 in / packed uint2 out (memory-bound pass).
// ---------------------------------------------------------------------------
__global__ void k0_cvt_bf16(const float* __restrict__ src,
                            unsigned short* __restrict__ dst, int n4) {
    int i = blockIdx.x * blockDim.x + threadIdx.x;
    if (i >= n4) return;
    float4 v = reinterpret_cast<const float4*>(src)[i];
    uint2 o;
    o.x = bfr(v.x) | (bfr(v.y) << 16);
    o.y = bfr(v.z) | (bfr(v.w) << 16);
    reinterpret_cast<uint2*>(dst)[i] = o;
}

// ---------------------------------------------------------------------------
// WMMA fragment helpers (CDNA5 lane layouts).  A fragment = two contiguous
// 8-element K-runs per lane -> two 16-byte global loads.
// ---------------------------------------------------------------------------
static __device__ __forceinline__ v16bf
load_a_bf16(const unsigned short* A, int lda, int row0, int k0, int lane) {
    int m = lane & 15, kh = lane >> 4;
    const unsigned short* p = A + (size_t)(row0 + m) * lda + k0 + kh * 16;
    union { uint4 q[2]; v16us us; } u;
    u.q[0] = *(const uint4*)(p);        // K = kh*16 + [0..8)
    u.q[1] = *(const uint4*)(p + 8);    // K = kh*16 + 8 + [0..8)
    return __builtin_bit_cast(v16bf, u.us);
}

// B fragment from k-major LDS tile [32][64] (read in kernel body via macro to
// keep addrspace(3) => ds_load)
#define FRAG_B_KM(dst, tb, cur, nbase)                                          \
    {                                                                           \
        v16us _u;                                                               \
        _Pragma("unroll")                                                       \
        for (int _i = 0; _i < 16; ++_i)                                         \
            _u[_i] = tb[cur][(((lane >> 4) * 16 + _i) << 6) + (nbase) + (lane & 15)]; \
        dst = __builtin_bit_cast(v16bf, _u);                                    \
    }

// B fragment from n-major LDS tile [64][32]: 16 contiguous ushorts per lane
#define FRAG_B_NM(dst, tb, cur, nbase)                                          \
    {                                                                           \
        union { uint4 q[2]; v16us us; } _u;                                     \
        int _o = (((nbase) + (lane & 15)) << 5) + ((lane >> 4) << 4);           \
        _u.q[0] = *(const uint4*)&tb[cur][_o];                                  \
        _u.q[1] = *(const uint4*)&tb[cur][_o + 8];                              \
        dst = __builtin_bit_cast(v16bf, _u.us);                                 \
    }

// --- double-buffered TDM staging of a 32(K) x 64(N) k-major B tile ----------
#if HAVE_TDM
#define B_KM_INIT(Bg, ldb, Ktot, colB)                                          \
    if (wid == 0)                                                               \
        tdm_load_2d_bf16(0u, (Bg) + (colB), 64, 32, (ldb), (Ktot), (ldb));
#define B_KM_STEP(Bg, ldb, Ktot, colB, s, S)                                    \
    {                                                                           \
        if (wid == 0) {                                                         \
            if ((s) + 1 < (S)) {                                                \
                tdm_load_2d_bf16((unsigned)((((s) + 1) & 1) * 4096),            \
                                 (Bg) + (size_t)(32 * ((s) + 1)) * (ldb) + (colB), \
                                 64, 32, (ldb), (Ktot), (ldb));                 \
                __builtin_amdgcn_s_wait_tensorcnt(1);                           \
            } else {                                                            \
                __builtin_amdgcn_s_wait_tensorcnt(0);                           \
            }                                                                   \
        }                                                                       \
        __syncthreads();                                                        \
        LDS_TOUCH();                                                            \
    }
#define B_NM_INIT(Vg, colB)                                                     \
    if (wid == 0)                                                               \
        tdm_load_2d_bf16(0u, (Vg) + (size_t)(colB) * NHW, 32, 64, NHW, NC, NHW);
#define B_NM_STEP(Vg, colB, s, S)                                               \
    {                                                                           \
        if (wid == 0) {                                                         \
            if ((s) + 1 < (S)) {                                                \
                tdm_load_2d_bf16((unsigned)((((s) + 1) & 1) * 4096),            \
                                 (Vg) + (size_t)(colB) * NHW + 32 * ((s) + 1),  \
                                 32, 64, NHW, NC, NHW);                         \
                __builtin_amdgcn_s_wait_tensorcnt(1);                           \
            } else {                                                            \
                __builtin_amdgcn_s_wait_tensorcnt(0);                           \
            }                                                                   \
        }                                                                       \
        __syncthreads();                                                        \
        LDS_TOUCH();                                                            \
    }
#define CUR(s) ((s) & 1)
#else
// Fallback: cooperative coalesced fill (two uint4 loads/stores per thread).
#define B_KM_INIT(Bg, ldb, Ktot, colB)
#define B_KM_STEP(Bg, ldb, Ktot, colB, s, S)                                    \
    {                                                                           \
        const unsigned short* _src = (Bg)                                       \
            + (size_t)(32 * (s) + (threadIdx.x >> 2)) * (ldb)                   \
            + (colB) + (threadIdx.x & 3) * 16;                                  \
        uint4 _q0 = ((const uint4*)_src)[0];                                    \
        uint4 _q1 = ((const uint4*)_src)[1];                                    \
        *(uint4*)&tb[0][threadIdx.x * 16]     = _q0;                            \
        *(uint4*)&tb[0][threadIdx.x * 16 + 8] = _q1;                            \
        __syncthreads();                                                        \
    }
#define B_NM_INIT(Vg, colB)
#define B_NM_STEP(Vg, colB, s, S)                                               \
    {                                                                           \
        const unsigned short* _src = (Vg)                                       \
            + (size_t)((colB) + (threadIdx.x >> 1)) * NHW                       \
            + 32 * (s) + (threadIdx.x & 1) * 16;                                \
        uint4 _q0 = ((const uint4*)_src)[0];                                    \
        uint4 _q1 = ((const uint4*)_src)[1];                                    \
        *(uint4*)&tb[0][threadIdx.x * 16]     = _q0;                            \
        *(uint4*)&tb[0][threadIdx.x * 16 + 8] = _q1;                            \
        __syncthreads();                                                        \
    }
#define CUR(s) 0
#endif

static __device__ __forceinline__ void
store_c_f32(const v8f& c, float* out, int ldo, int row0, int col0, int lane) {
    int n = lane & 15, mh = lane >> 4;
#pragma unroll
    for (int r = 0; r < 8; ++r)
        out[(size_t)(row0 + r + mh * 8) * ldo + col0 + n] = c[r];
}

static __device__ __forceinline__ void
store_c_relu_f32(const v8f& c, float* out, int ldo, int row0, int col0,
                 const float* bias, int lane) {
    int n = lane & 15, mh = lane >> 4;
    float bv = bias[col0 + n];
#pragma unroll
    for (int r = 0; r < 8; ++r)
        out[(size_t)(row0 + r + mh * 8) * ldo + col0 + n] = fmaxf(c[r] + bv, 0.f);
}

static __device__ __forceinline__ void
store_c_relu_bf16(const v8f& c, unsigned short* out, int ldo, int row0, int col0,
                  const float* bias, int lane) {
    int n = lane & 15, mh = lane >> 4;
    float bv = bias[col0 + n];
#pragma unroll
    for (int r = 0; r < 8; ++r)
        out[(size_t)(row0 + r + mh * 8) * ldo + col0 + n] = f2bf(fmaxf(c[r] + bv, 0.f));
}

static __device__ __forceinline__ void
store_c_bf16(const v8f& c, unsigned short* out, int ldo, int row0, int col0, int lane) {
    int n = lane & 15, mh = lane >> 4;
#pragma unroll
    for (int r = 0; r < 8; ++r)
        out[(size_t)(row0 + r + mh * 8) * ldo + col0 + n] = f2bf(c[r]);
}

#define WMMA_BF16(a, b, c) \
    __builtin_amdgcn_wmma_f32_16x16x32_bf16(false, (a), false, (b), (short)0, (c), false, false)

// ---------------------------------------------------------------------------
// K1: q = ReLU(obj_bf @ Wobj_bf + b_obj).  grid (8,32), block 128.
// ---------------------------------------------------------------------------
__global__ void k1_obj_gemm(const unsigned short* __restrict__ obj_bf,
                            const unsigned short* __restrict__ Wo_bf,
                            const float* __restrict__ bo,
                            unsigned short* __restrict__ q_bf) {
    __shared__ unsigned short tb[2][32 * 64];
    LDS_SEED(tb, 2047);
    int lane = threadIdx.x & 31, wid = threadIdx.x >> 5;
    int colB = blockIdx.x * 64;
    int row0 = blockIdx.y * 16;
    const int S = DIN / 32;
    v8f c = {};
    B_KM_INIT(Wo_bf, NC, DIN, colB);
    for (int s = 0; s < S; ++s) {
        B_KM_STEP(Wo_bf, NC, DIN, colB, s, S);
        v16bf a = load_a_bf16(obj_bf, DIN, row0, 32 * s, lane);
        v16bf bb; FRAG_B_KM(bb, tb, CUR(s), wid * 16);
        c = WMMA_BF16(a, bb, c);
        __syncthreads();
    }
    store_c_relu_bf16(c, q_bf, NC, row0, colB + wid * 16, bo, lane);
}

// ---------------------------------------------------------------------------
// K2: dots[b] = q[b] @ key[b]  (16 x 512 x 2304).  grid (36,32), block 128.
// ---------------------------------------------------------------------------
__global__ void k2_dots(const unsigned short* __restrict__ q_bf,
                        const unsigned short* __restrict__ key_bf,
                        float* __restrict__ dots) {
    __shared__ unsigned short tb[2][32 * 64];
    LDS_SEED(tb, 2047);
    int lane = threadIdx.x & 31, wid = threadIdx.x >> 5;
    int colB = blockIdx.x * 64;
    int b = blockIdx.y;
    const unsigned short* A  = q_bf   + (size_t)b * NK * NC;
    const unsigned short* Bg = key_bf + (size_t)b * NC * NHW;
    const int S = NC / 32;
    v8f c = {};
    B_KM_INIT(Bg, NHW, NC, colB);
    for (int s = 0; s < S; ++s) {
        B_KM_STEP(Bg, NHW, NC, colB, s, S);
        v16bf a = load_a_bf16(A, NC, 0, 32 * s, lane);
        v16bf bb; FRAG_B_KM(bb, tb, CUR(s), wid * 16);
        c = WMMA_BF16(a, bb, c);
        __syncthreads();
    }
    store_c_f32(c, dots + (size_t)b * NK * NHW, NHW, 0, colB + wid * 16, lane);
}

// ---------------------------------------------------------------------------
// K3: row softmax over HW=2304, one wave32 per row -> bf16 attn.
// ---------------------------------------------------------------------------
__global__ void k3_softmax(const float* __restrict__ dots,
                           unsigned short* __restrict__ attn) {
    int row = blockIdx.x;
    int lane = threadIdx.x;
    const float* p = dots + (size_t)row * NHW;
    unsigned short* o = attn + (size_t)row * NHW;

    float mx = -INFINITY;
    for (int h = lane; h < NHW; h += 32) mx = fmaxf(mx, p[h]);
#pragma unroll
    for (int off = 16; off >= 1; off >>= 1) mx = fmaxf(mx, __shfl_xor(mx, off, 32));

    float sum = 0.f;
    for (int h = lane; h < NHW; h += 32) sum += __expf(p[h] - mx);
#pragma unroll
    for (int off = 16; off >= 1; off >>= 1) sum += __shfl_xor(sum, off, 32);
    float inv = 1.f / sum;

    for (int h = lane; h < NHW; h += 32) o[h] = f2bf(__expf(p[h] - mx) * inv);
}

// ---------------------------------------------------------------------------
// K4: ctx[b] = attn[b] @ val[b]^T  (16 x 2304 x 512).  grid (8,32), block 128.
// Val tile staged n-major (64 channel rows x 32 contiguous hw).
// ---------------------------------------------------------------------------
__global__ void k4_ctx(const unsigned short* __restrict__ attn,
                       const unsigned short* __restrict__ val_bf,
                       unsigned short* __restrict__ ctx_bf) {
    __shared__ unsigned short tb[2][64 * 32];
    LDS_SEED(tb, 2047);
    int lane = threadIdx.x & 31, wid = threadIdx.x >> 5;
    int colB = blockIdx.x * 64;
    int b = blockIdx.y;
    const unsigned short* A  = attn   + (size_t)b * NK * NHW;
    const unsigned short* Vg = val_bf + (size_t)b * NC * NHW;
    const int S = NHW / 32;
    v8f c = {};
    B_NM_INIT(Vg, colB);
    for (int s = 0; s < S; ++s) {
        B_NM_STEP(Vg, colB, s, S);
        v16bf a = load_a_bf16(A, NHW, 0, 32 * s, lane);
        v16bf bb; FRAG_B_NM(bb, tb, CUR(s), wid * 16);
        c = WMMA_BF16(a, bb, c);
        __syncthreads();
    }
    store_c_bf16(c, ctx_bf, NC, b * NK, colB + wid * 16, lane);
}

// ---------------------------------------------------------------------------
// K5: ctx_t = ReLU(ctx @ W_ctx + b_ctx)  (512 x 512 x 1024) -> bf16.
// grid (16,32), block 128.
// ---------------------------------------------------------------------------
__global__ void k5_ctxlin(const unsigned short* __restrict__ ctx_bf,
                          const unsigned short* __restrict__ Wc_bf,
                          const float* __restrict__ bc,
                          unsigned short* __restrict__ ctxt_bf) {
    __shared__ unsigned short tb[2][32 * 64];
    LDS_SEED(tb, 2047);
    int lane = threadIdx.x & 31, wid = threadIdx.x >> 5;
    int colB = blockIdx.x * 64;
    int row0 = blockIdx.y * 16;
    const int S = NC / 32;
    v8f c = {};
    B_KM_INIT(Wc_bf, DOUT, NC, colB);
    for (int s = 0; s < S; ++s) {
        B_KM_STEP(Wc_bf, DOUT, NC, colB, s, S);
        v16bf a = load_a_bf16(ctx_bf, NC, row0, 32 * s, lane);
        v16bf bb; FRAG_B_KM(bb, tb, CUR(s), wid * 16);
        c = WMMA_BF16(a, bb, c);
        __syncthreads();
    }
    store_c_relu_bf16(c, ctxt_bf, DOUT, row0, colB + wid * 16, bc, lane);
}

// ---------------------------------------------------------------------------
// K6: feats = ReLU([obj_feat, ctx_t] @ W_cat + b_cat)  (512 x 3072 x 1024)
// Split-K A source; B staged via TDM.  grid (16,32), block 128.
// ---------------------------------------------------------------------------
__global__ void k6_cat(const unsigned short* __restrict__ obj_bf,
                       const unsigned short* __restrict__ ctxt_bf,
                       const unsigned short* __restrict__ Wcat_bf,
                       const float* __restrict__ bcat,
                       float* __restrict__ feats) {
    __shared__ unsigned short tb[2][32 * 64];
    LDS_SEED(tb, 2047);
    int lane = threadIdx.x & 31, wid = threadIdx.x >> 5;
    int colB = blockIdx.x * 64;
    int row0 = blockIdx.y * 16;
    const int S = DCAT / 32;
    v8f c = {};
    B_KM_INIT(Wcat_bf, DOUT, DCAT, colB);
    for (int s = 0; s < S; ++s) {
        B_KM_STEP(Wcat_bf, DOUT, DCAT, colB, s, S);
        int k0 = 32 * s;
        v16bf a = (k0 < DIN)
                      ? load_a_bf16(obj_bf, DIN, row0, k0, lane)
                      : load_a_bf16(ctxt_bf, DOUT, row0, k0 - DIN, lane);
        v16bf bb; FRAG_B_KM(bb, tb, CUR(s), wid * 16);
        c = WMMA_BF16(a, bb, c);
        __syncthreads();
    }
    store_c_relu_f32(c, feats, DOUT, row0, colB + wid * 16, bcat, lane);
}

// ---------------------------------------------------------------------------
// K7: paired = 0.5*(feats[b,p0] + feats[b,p1]) -> out [800, 1024] fp32.
// ---------------------------------------------------------------------------
__global__ void k7_pairs(const float* __restrict__ feats,
                         const int* __restrict__ pairs,
                         float* __restrict__ out) {
    int br = blockIdx.x;
    int b = br / NR;
    int p0 = pairs[(size_t)br * 2 + 0];
    int p1 = pairs[(size_t)br * 2 + 1];
    const float* f0 = feats + (size_t)(b * NK + p0) * DOUT;
    const float* f1 = feats + (size_t)(b * NK + p1) * DOUT;
    float* o = out + (size_t)br * DOUT;
    for (int i = threadIdx.x; i < DOUT; i += blockDim.x)
        o[i] = 0.5f * (f0[i] + f1[i]);
}

// ---------------------------------------------------------------------------
// Launcher
// ---------------------------------------------------------------------------
extern "C" void kernel_launch(void* const* d_in, const int* in_sizes, int n_in,
                              void* d_out, int out_size, void* d_ws, size_t ws_size,
                              hipStream_t stream) {
    const float* obj_feat = (const float*)d_in[0];
    const float* ctx_key  = (const float*)d_in[1];
    const float* ctx_val  = (const float*)d_in[2];
    const int*   pairs    = (const int*)d_in[4];
    const float* W_obj    = (const float*)d_in[7];
    const float* b_obj    = (const float*)d_in[8];
    const float* W_ctx    = (const float*)d_in[9];
    const float* b_ctx    = (const float*)d_in[10];
    const float* W_cat    = (const float*)d_in[11];
    const float* b_cat    = (const float*)d_in[12];
    float* out = (float*)d_out;

    char* ws = (char*)d_ws;
    size_t off = 0;
    unsigned short* key_bf  = (unsigned short*)(ws + off); off += (size_t)NB * NC * NHW * 2;
    unsigned short* val_bf  = (unsigned short*)(ws + off); off += (size_t)NB * NC * NHW * 2;
    unsigned short* obj_bf  = (unsigned short*)(ws + off); off += (size_t)NOBJ * DIN * 2;
    unsigned short* Wo_bf   = (unsigned short*)(ws + off); off += (size_t)DIN * NC * 2;
    unsigned short* Wc_bf   = (unsigned short*)(ws + off); off += (size_t)NC * DOUT * 2;
    unsigned short* Wcat_bf = (unsigned short*)(ws + off); off += (size_t)DCAT * DOUT * 2;
    unsigned short* q_bf    = (unsigned short*)(ws + off); off += (size_t)NOBJ * NC * 2;
    float*          dots    = (float*)(ws + off);          off += (size_t)NOBJ * NHW * 4;
    unsigned short* attn    = (unsigned short*)(ws + off); off += (size_t)NOBJ * NHW * 2;
    unsigned short* ctx_bf  = (unsigned short*)(ws + off); off += (size_t)NOBJ * NC * 2;
    unsigned short* ctxt_bf = (unsigned short*)(ws + off); off += (size_t)NOBJ * DOUT * 2;
    float*          feats   = (float*)(ws + off);          off += (size_t)NOBJ * DOUT * 4;
    (void)ws_size; (void)in_sizes; (void)n_in; (void)out_size;

    auto cvt = [&](const float* s, unsigned short* d, size_t n) {
        int n4 = (int)(n / 4);
        k0_cvt_bf16<<<dim3((n4 + 255) / 256), dim3(256), 0, stream>>>(s, d, n4);
    };
    cvt(ctx_key,  key_bf,  (size_t)NB * NC * NHW);
    cvt(ctx_val,  val_bf,  (size_t)NB * NC * NHW);
    cvt(obj_feat, obj_bf,  (size_t)NOBJ * DIN);
    cvt(W_obj,    Wo_bf,   (size_t)DIN * NC);
    cvt(W_ctx,    Wc_bf,   (size_t)NC * DOUT);
    cvt(W_cat,    Wcat_bf, (size_t)DCAT * DOUT);

    dim3 blk(128);
    k1_obj_gemm<<<dim3(NC / 64, NOBJ / 16), blk, 0, stream>>>(obj_bf, Wo_bf, b_obj, q_bf);
    k2_dots   <<<dim3(NHW / 64, NB), blk, 0, stream>>>(q_bf, key_bf, dots);
    k3_softmax<<<dim3(NOBJ), dim3(32), 0, stream>>>(dots, attn);
    k4_ctx    <<<dim3(NC / 64, NB), blk, 0, stream>>>(attn, val_bf, ctx_bf);
    k5_ctxlin <<<dim3(DOUT / 64, NOBJ / 16), blk, 0, stream>>>(ctx_bf, Wc_bf, b_ctx, ctxt_bf);
    k6_cat    <<<dim3(DOUT / 64, NOBJ / 16), blk, 0, stream>>>(obj_bf, ctxt_bf, Wcat_bf, b_cat, feats);
    k7_pairs  <<<dim3(NB * NR), dim3(256), 0, stream>>>(feats, pairs, out);
}
